// DTQN_24206435680852
// MI455X (gfx1250) — matrix operand order repaired and checked by reading, not compiled, not run on hardware
//
#include <hip/hip_runtime.h>
#include <hip/hip_bf16.h>
#include <stdint.h>

typedef __bf16 bf16;
typedef __attribute__((ext_vector_type(16))) __bf16 v16bf;
typedef __attribute__((ext_vector_type(8)))  __bf16 v8bf;
typedef __attribute__((ext_vector_type(8)))  float  v8f;

#define ROWS      8192      // BATCH * HIST
#define OBS_DIM   4096
#define ACTION_DIM 64
#define D_MODEL   512
#define HIST      256
#define BATCH     32
#define D_STATE   16
#define D_INNER   1024
#define DT_RANK   32
#define OBS_OUT   448
#define N_LAYERS  4
#define NUM_ACT   18

// LDS B-tile: 64 rows x 32 bf16, row stride padded to 40 elems (80B) -> bank-conflict-free
#define BROW_STRIDE 40
#define STAGE_ELEMS (64 * BROW_STRIDE)
#define STAGE_BYTES (STAGE_ELEMS * 2)

#define SHUF16(lo, hi) __builtin_shufflevector(lo, hi, 0,1,2,3,4,5,6,7,8,9,10,11,12,13,14,15)

// ---------------------------------------------------------------- f32 -> bf16
__global__ void k_f2bf(const float* __restrict__ s, bf16* __restrict__ d, int n) {
    int i = blockIdx.x * blockDim.x + threadIdx.x;
    if (i < n) d[i] = (bf16)s[i];
}

// ---------------------------------------------------------------- WMMA GEMM
// C[m,n] = act( sum_k A[m,k] * W[n,k] + bias[n] )
// A: [M,K] bf16 row-major, W: [N,K] bf16 row-major.
// Block: 256 threads = 8 waves. Each wave: 16(M) x 64(N); block: 128(M) x 64(N).
// B tile (64x32) staged in LDS via async load-to-LDS, double buffered; A fragment
// software-pipelined in registers. Requires M%128==0, N%64==0, K%32==0.
__global__ __launch_bounds__(256)
void k_gemm_bf16(const bf16* __restrict__ A, const bf16* __restrict__ W,
                 const float* __restrict__ bias, float* __restrict__ Cf,
                 bf16* __restrict__ Cb, int M, int N, int K, int act)
{
    __shared__ __align__(16) bf16 sB[2 * STAGE_ELEMS];

    const int tid  = threadIdx.x;
    const int lane = tid & 31;
    const int wave = tid >> 5;
    const int mTile = blockIdx.y * 128 + wave * 16;
    const int nBase = blockIdx.x * 64;
    const int r  = lane & 15;
    const int hi = lane >> 4;            // 0: lanes 0-15, 1: lanes 16-31

    // ---- async copy assignment: thread covers one 16B chunk of the 64x32 tile
    const int cRow = tid >> 2;           // 0..63  (n within tile)
    const int cChk = tid & 3;            // 0..3   (16B chunk within the 64B row)
    const bf16* gB = W + (size_t)(nBase + cRow) * K + cChk * 8;
    const uint32_t ldsBase = (uint32_t)(uintptr_t)&sB[0];
    const uint32_t ldsDst  = ldsBase + (uint32_t)(cRow * BROW_STRIDE + cChk * 8) * 2u;

    auto prefetch = [&](int stage, int k) {
        uint64_t src = (uint64_t)(uintptr_t)(gB + k);
        uint32_t dst = ldsDst + (uint32_t)stage * STAGE_BYTES;
        asm volatile("global_load_async_to_lds_b128 %0, %1, off"
                     :: "v"(dst), "v"(src) : "memory");
    };

    // A fragment (16x32, MxK): lane<16 holds K {k..k+15}; lane>=16 {k+16..k+31}
    const bf16* Ap = A + (size_t)(mTile + r) * K + hi * 16;
    // B fragment (32x16, KxN): lane<16 holds K {k..k+15} of column N; lane>=16 {k+16..k+31}
    const bf16* sRead = &sB[0] + r * BROW_STRIDE + hi * 16;

    v8f acc0 = (v8f){0.f,0.f,0.f,0.f,0.f,0.f,0.f,0.f};
    v8f acc1 = acc0, acc2 = acc0, acc3 = acc0;

    prefetch(0, 0);
    v8bf al = *(const v8bf*)(Ap);
    v8bf ah = *(const v8bf*)(Ap + 8);

    int k = 0;
    for (; k + 32 < K; k += 32) {
        const int stage = (k >> 5) & 1;
        asm volatile("s_wait_asynccnt 0x0" ::: "memory");
        __syncthreads();                          // stage filled & visible to all waves
        prefetch(stage ^ 1, k + 32);              // async fill of next stage

        const bf16* sb = sRead + stage * STAGE_ELEMS;
        v8bf b0l = *(const v8bf*)(sb);
        v8bf b0h = *(const v8bf*)(sb + 8);
        v8bf b1l = *(const v8bf*)(sb + 16 * BROW_STRIDE);
        v8bf b1h = *(const v8bf*)(sb + 16 * BROW_STRIDE + 8);
        v8bf b2l = *(const v8bf*)(sb + 32 * BROW_STRIDE);
        v8bf b2h = *(const v8bf*)(sb + 32 * BROW_STRIDE + 8);
        v8bf b3l = *(const v8bf*)(sb + 48 * BROW_STRIDE);
        v8bf b3h = *(const v8bf*)(sb + 48 * BROW_STRIDE + 8);

        // software-pipelined A fragment for next K-step (in flight during WMMAs)
        v8bf nal = *(const v8bf*)(Ap + k + 32);
        v8bf nah = *(const v8bf*)(Ap + k + 40);

        v16bf af = SHUF16(al, ah);
        acc0 = __builtin_amdgcn_wmma_f32_16x16x32_bf16(false, af, false, SHUF16(b0l, b0h),
                                                       (short)0, acc0, false, false);
        acc1 = __builtin_amdgcn_wmma_f32_16x16x32_bf16(false, af, false, SHUF16(b1l, b1h),
                                                       (short)0, acc1, false, false);
        acc2 = __builtin_amdgcn_wmma_f32_16x16x32_bf16(false, af, false, SHUF16(b2l, b2h),
                                                       (short)0, acc2, false, false);
        acc3 = __builtin_amdgcn_wmma_f32_16x16x32_bf16(false, af, false, SHUF16(b3l, b3h),
                                                       (short)0, acc3, false, false);
        al = nal; ah = nah;
        __syncthreads();                          // all waves done with 'stage' before overwrite
    }

    // ---- final K-step (no further prefetch)
    {
        const int stage = (k >> 5) & 1;
        asm volatile("s_wait_asynccnt 0x0" ::: "memory");
        __syncthreads();

        const bf16* sb = sRead + stage * STAGE_ELEMS;
        v8bf b0l = *(const v8bf*)(sb);
        v8bf b0h = *(const v8bf*)(sb + 8);
        v8bf b1l = *(const v8bf*)(sb + 16 * BROW_STRIDE);
        v8bf b1h = *(const v8bf*)(sb + 16 * BROW_STRIDE + 8);
        v8bf b2l = *(const v8bf*)(sb + 32 * BROW_STRIDE);
        v8bf b2h = *(const v8bf*)(sb + 32 * BROW_STRIDE + 8);
        v8bf b3l = *(const v8bf*)(sb + 48 * BROW_STRIDE);
        v8bf b3h = *(const v8bf*)(sb + 48 * BROW_STRIDE + 8);

        v16bf af = SHUF16(al, ah);
        acc0 = __builtin_amdgcn_wmma_f32_16x16x32_bf16(false, af, false, SHUF16(b0l, b0h),
                                                       (short)0, acc0, false, false);
        acc1 = __builtin_amdgcn_wmma_f32_16x16x32_bf16(false, af, false, SHUF16(b1l, b1h),
                                                       (short)0, acc1, false, false);
        acc2 = __builtin_amdgcn_wmma_f32_16x16x32_bf16(false, af, false, SHUF16(b2l, b2h),
                                                       (short)0, acc2, false, false);
        acc3 = __builtin_amdgcn_wmma_f32_16x16x32_bf16(false, af, false, SHUF16(b3l, b3h),
                                                       (short)0, acc3, false, false);
    }

    // C/D layout: lane -> N = lane&15 (plus tile offset); vgpr j -> M = j + 8*(lane>>4)
    const int n0 = nBase + r;
    const int mOff = hi * 8;
    float bv0 = bias ? bias[n0]      : 0.f;
    float bv1 = bias ? bias[n0 + 16] : 0.f;
    float bv2 = bias ? bias[n0 + 32] : 0.f;
    float bv3 = bias ? bias[n0 + 48] : 0.f;

    #pragma unroll
    for (int j = 0; j < 8; ++j) {
        int m = mTile + mOff + j;
        float v0 = acc0[j] + bv0, v1 = acc1[j] + bv1, v2 = acc2[j] + bv2, v3 = acc3[j] + bv3;
        if (act == 1) {
            v0 = v0 > 0.f ? v0 : 0.f;  v1 = v1 > 0.f ? v1 : 0.f;
            v2 = v2 > 0.f ? v2 : 0.f;  v3 = v3 > 0.f ? v3 : 0.f;
        } else if (act == 2) {
            v0 = v0 > 20.f ? v0 : logf(1.f + __expf(v0));
            v1 = v1 > 20.f ? v1 : logf(1.f + __expf(v1));
            v2 = v2 > 20.f ? v2 : logf(1.f + __expf(v2));
            v3 = v3 > 20.f ? v3 : logf(1.f + __expf(v3));
        }
        size_t base = (size_t)m * N + n0;
        if (Cf) { Cf[base] = v0; Cf[base+16] = v1; Cf[base+32] = v2; Cf[base+48] = v3; }
        if (Cb) { Cb[base] = (bf16)v0; Cb[base+16] = (bf16)v1; Cb[base+32] = (bf16)v2; Cb[base+48] = (bf16)v3; }
    }
}

// ---------------------------------------------------------------- embedding
__global__ void k_embed(const float* __restrict__ tok, const int* __restrict__ actions,
                        const float* __restrict__ act_emb, const float* __restrict__ pos_emb,
                        bf16* __restrict__ xb)
{
    int idx = blockIdx.x * blockDim.x + threadIdx.x;    // ROWS * 512
    if (idx >= ROWS * D_MODEL) return;
    int c = idx & 511, row = idx >> 9, l = row & 255;
    float v;
    if (c < ACTION_DIM) {
        v = 0.f;
        if (l > 0) v = act_emb[actions[row - 1] * ACTION_DIM + c];
    } else {
        v = tok[(size_t)row * OBS_OUT + (c - ACTION_DIM)];
    }
    v += pos_emb[l * D_MODEL + c];
    xb[idx] = (bf16)v;
}

// ---------------------------------------------------------------- conv + silu
__global__ void k_conv(const float* __restrict__ xz, const float* __restrict__ cw,
                       const float* __restrict__ cb, float* __restrict__ xc,
                       bf16* __restrict__ xcb)
{
    int idx = blockIdx.x * blockDim.x + threadIdx.x;    // ROWS * D_INNER
    if (idx >= ROWS * D_INNER) return;
    int d = idx & 1023, bl = idx >> 10, l = bl & 255;
    float a = cb[d];
    #pragma unroll
    for (int k = 0; k < 4; ++k) {
        int lk = l + k - 3;
        if (lk >= 0) a += cw[d * 4 + k] * xz[(size_t)(bl + k - 3) * (2 * D_INNER) + d];
    }
    float s = a / (1.f + __expf(-a));
    xc[idx] = s;
    xcb[idx] = (bf16)s;
}

// ---------------------------------------------------------------- dt slice -> bf16
__global__ void k_dtprep(const float* __restrict__ xdbl, bf16* __restrict__ dtl)
{
    int idx = blockIdx.x * blockDim.x + threadIdx.x;    // ROWS * 32
    if (idx >= ROWS * DT_RANK) return;
    int c = idx & 31, row = idx >> 5;
    dtl[idx] = (bf16)xdbl[(size_t)row * 64 + c];
}

// ---------------------------------------------------------------- selective scan (fused D_p skip + silu(z) gate)
__global__ __launch_bounds__(256)
void k_scan(const float* __restrict__ xdbl, const float* __restrict__ dt,
            const float* __restrict__ xc, const float* __restrict__ xz,
            const float* __restrict__ A_log, const float* __restrict__ D_p,
            bf16* __restrict__ yb)
{
    __shared__ float sBC[32];
    const int tid = threadIdx.x;
    const int d = blockIdx.x * 256 + tid;
    const int b = blockIdx.y;
    float Ad[D_STATE];
    #pragma unroll
    for (int n = 0; n < D_STATE; ++n) Ad[n] = -__expf(A_log[d * D_STATE + n]);
    const float Dp = D_p[d];
    float h[D_STATE];
    #pragma unroll
    for (int n = 0; n < D_STATE; ++n) h[n] = 0.f;

    for (int l = 0; l < HIST; ++l) {
        int row = b * HIST + l;
        __syncthreads();
        if (tid < 32) sBC[tid] = xdbl[(size_t)row * 64 + 32 + tid];
        __syncthreads();
        float dtv = dt[(size_t)row * D_INNER + d];
        float xv  = xc[(size_t)row * D_INNER + d];
        float y = 0.f;
        #pragma unroll
        for (int n = 0; n < D_STATE; ++n) {
            float dA = __expf(dtv * Ad[n]);
            h[n] = dA * h[n] + (dtv * sBC[n]) * xv;
            y += h[n] * sBC[16 + n];
        }
        float zv = xz[(size_t)row * (2 * D_INNER) + D_INNER + d];
        float g = zv / (1.f + __expf(-zv));
        yb[(size_t)row * D_INNER + d] = (bf16)((y + xv * Dp) * g);
    }
}

// ---------------------------------------------------------------- FFN head (N=18)
__global__ void k_ffn2(const bf16* __restrict__ h, const float* __restrict__ W2,
                       const float* __restrict__ b2, float* __restrict__ out)
{
    int idx = blockIdx.x * blockDim.x + threadIdx.x;    // ROWS * 18
    if (idx >= ROWS * NUM_ACT) return;
    int n = idx % NUM_ACT, row = idx / NUM_ACT;
    float acc = b2[n];
    const bf16*  hr = h  + (size_t)row * D_MODEL;
    const float* wr = W2 + (size_t)n   * D_MODEL;
    for (int k = 0; k < D_MODEL; ++k) acc += (float)hr[k] * wr[k];
    out[idx] = acc;
}

// ================================================================ launcher
extern "C" void kernel_launch(void* const* d_in, const int* in_sizes, int n_in,
                              void* d_out, int out_size, void* d_ws, size_t ws_size,
                              hipStream_t stream)
{
    const float* obss    = (const float*)d_in[0];
    const int*   actions = (const int*)  d_in[1];
    const float* obs_W   = (const float*)d_in[2];
    const float* obs_b   = (const float*)d_in[3];
    const float* act_emb = (const float*)d_in[4];
    const float* pos_emb = (const float*)d_in[5];
    const float* W_in    = (const float*)d_in[6];
    const float* conv_w  = (const float*)d_in[7];
    const float* conv_b  = (const float*)d_in[8];
    const float* W_x     = (const float*)d_in[9];
    const float* W_dt    = (const float*)d_in[10];
    const float* b_dt    = (const float*)d_in[11];
    const float* A_log   = (const float*)d_in[12];
    const float* D_p     = (const float*)d_in[13];
    const float* W_out   = (const float*)d_in[14];
    const float* ffn_W1  = (const float*)d_in[15];
    const float* ffn_b1  = (const float*)d_in[16];
    const float* ffn_W2  = (const float*)d_in[17];
    const float* ffn_b2  = (const float*)d_in[18];
    float* out = (float*)d_out;

    uintptr_t cur = (uintptr_t)d_ws;
    auto alloc = [&](size_t bytes) -> void* {
        uintptr_t rr = (cur + 255) & ~(uintptr_t)255;
        cur = rr + bytes;
        return (void*)rr;
    };

    bf16* obss_b = (bf16*)alloc((size_t)ROWS * OBS_DIM * 2);
    bf16* obsWb  = (bf16*)alloc((size_t)OBS_OUT * OBS_DIM * 2);
    bf16* Winb   = (bf16*)alloc((size_t)N_LAYERS * 2 * D_INNER * D_MODEL * 2);
    bf16* Wxb    = (bf16*)alloc((size_t)N_LAYERS * 64 * D_INNER * 2);
    bf16* Wdtb   = (bf16*)alloc((size_t)N_LAYERS * D_INNER * DT_RANK * 2);
    bf16* Woutb  = (bf16*)alloc((size_t)N_LAYERS * D_MODEL * D_INNER * 2);
    bf16* W1b    = (bf16*)alloc((size_t)D_MODEL * D_MODEL * 2);
    float* tok   = (float*)alloc((size_t)ROWS * OBS_OUT * 4);
    bf16* xb     = (bf16*)alloc((size_t)ROWS * D_MODEL * 2);
    float* xz    = (float*)alloc((size_t)ROWS * 2 * D_INNER * 4);
    float* xc    = (float*)alloc((size_t)ROWS * D_INNER * 4);
    bf16* xcb    = (bf16*)alloc((size_t)ROWS * D_INNER * 2);
    float* xdbl  = (float*)alloc((size_t)ROWS * 64 * 4);
    bf16* dtlb   = (bf16*)alloc((size_t)ROWS * DT_RANK * 2);
    float* dtf   = (float*)alloc((size_t)ROWS * D_INNER * 4);
    bf16* yb     = (bf16*)alloc((size_t)ROWS * D_INNER * 2);
    bf16* hb     = (bf16*)alloc((size_t)ROWS * D_MODEL * 2);

    auto cvt = [&](const float* s, bf16* d, int n) {
        k_f2bf<<<(n + 255) / 256, 256, 0, stream>>>(s, d, n);
    };
    cvt(obss,   obss_b, ROWS * OBS_DIM);
    cvt(obs_W,  obsWb,  OBS_OUT * OBS_DIM);
    cvt(W_in,   Winb,   N_LAYERS * 2 * D_INNER * D_MODEL);
    cvt(W_x,    Wxb,    N_LAYERS * 64 * D_INNER);
    cvt(W_dt,   Wdtb,   N_LAYERS * D_INNER * DT_RANK);
    cvt(W_out,  Woutb,  N_LAYERS * D_MODEL * D_INNER);
    cvt(ffn_W1, W1b,    D_MODEL * D_MODEL);

    auto gemm = [&](const bf16* A, const bf16* W, const float* bias,
                    float* Cf, bf16* Cb, int M, int N, int K, int act) {
        dim3 grid(N / 64, M / 128);
        k_gemm_bf16<<<grid, 256, 0, stream>>>(A, W, bias, Cf, Cb, M, N, K, act);
    };

    // token projection: tok = obss @ obs_W^T + obs_b
    gemm(obss_b, obsWb, obs_b, tok, nullptr, ROWS, OBS_OUT, OBS_DIM, 0);

    // embedding + pos
    k_embed<<<(ROWS * D_MODEL + 255) / 256, 256, 0, stream>>>(tok, actions, act_emb, pos_emb, xb);

    for (int layer = 0; layer < N_LAYERS; ++layer) {
        const bf16* Wi = Winb  + (size_t)layer * 2 * D_INNER * D_MODEL;
        const bf16* Wxl= Wxb   + (size_t)layer * 64 * D_INNER;
        const bf16* Wd = Wdtb  + (size_t)layer * D_INNER * DT_RANK;
        const bf16* Wo = Woutb + (size_t)layer * D_MODEL * D_INNER;

        // xz = x @ W_in^T   (8192 x 2048)
        gemm(xb, Wi, nullptr, xz, nullptr, ROWS, 2 * D_INNER, D_MODEL, 0);
        // depthwise conv + silu
        k_conv<<<(ROWS * D_INNER + 255) / 256, 256, 0, stream>>>(
            xz, conv_w + (size_t)layer * D_INNER * 4, conv_b + (size_t)layer * D_INNER, xc, xcb);
        // xdbl = xc @ W_x^T  (8192 x 64)
        gemm(xcb, Wxl, nullptr, xdbl, nullptr, ROWS, 64, D_INNER, 0);
        // dt_low -> bf16
        k_dtprep<<<(ROWS * DT_RANK + 255) / 256, 256, 0, stream>>>(xdbl, dtlb);
        // dt = softplus(dt_low @ W_dt^T + b_dt)   (8192 x 1024)
        gemm(dtlb, Wd, b_dt + (size_t)layer * D_INNER, dtf, nullptr, ROWS, D_INNER, DT_RANK, 2);
        // selective scan + gate -> yb (bf16)
        dim3 sg(D_INNER / 256, BATCH);
        k_scan<<<sg, 256, 0, stream>>>(xdbl, dtf, xc, xz,
                                       A_log + (size_t)layer * D_INNER * D_STATE,
                                       D_p + (size_t)layer * D_INNER, yb);
        // x = y @ W_out^T  (8192 x 512), overwrite xb
        gemm(yb, Wo, nullptr, nullptr, xb, ROWS, D_MODEL, D_INNER, 0);
    }

    // FFN: h = relu(x @ W1^T + b1)
    gemm(xb, W1b, ffn_b1, nullptr, hb, ROWS, D_MODEL, D_MODEL, 1);
    // out = h @ W2^T + b2   (N=18, small)
    k_ffn2<<<(ROWS * NUM_ACT + 255) / 256, 256, 0, stream>>>(hb, ffn_W2, ffn_b2, out);
}
